// CausalMasklessSelfAttention_46196668236225
// MI455X (gfx1250) — matrix operand order repaired and checked by reading, not compiled, not run on hardware
//
#include <hip/hip_runtime.h>
#include <stdint.h>

typedef __bf16 bf16_t;
typedef __attribute__((ext_vector_type(8)))  __bf16 v8bf;
typedef __attribute__((ext_vector_type(16))) __bf16 v16bf;
typedef __attribute__((ext_vector_type(8)))  float  v8f;

#define B_  2
#define L_  2048
#define C_  1024
#define H_  16
#define D_  64
#define M_  (B_ * L_)   // 4096

// ---------------------------------------------------------------------------
// CDNA5 async global->LDS (ASYNCcnt path, bypasses VGPRs).
// LDS address comes from the VDST VGPR; the LDS offset is the low 32 bits of the
// generic pointer (shared aperture keeps offset in addr[31:0]).
// ---------------------------------------------------------------------------
__device__ inline void async_g2l_b128(void* lds, const void* g) {
  asm volatile("global_load_async_to_lds_b128 %0, %1, off"
               :: "v"((unsigned)(uintptr_t)lds), "v"(g) : "memory");
}
__device__ inline void wait_async0() {
  asm volatile("s_wait_asynccnt 0x0" ::: "memory");
}

// ---------------------------------------------------------------------------
// Fragment loaders (LDS -> VGPR) per CDNA5 ISA 7.12.2 16-bit layouts.
// ---------------------------------------------------------------------------
__device__ inline v16bf ld_a_frag(const bf16_t* base, int lda, int lane) {
  const int r = lane & 15, h = (lane >> 4) & 1;
  const bf16_t* p = base + r * lda + h * 8;
  v8bf lo = *(const v8bf*)(p);
  v8bf hi = *(const v8bf*)(p + 16);
  v16bf out;
#pragma unroll
  for (int i = 0; i < 8; ++i) { out[i] = lo[i]; out[8 + i] = hi[i]; }
  return out;
}

__device__ inline v16bf ld_b_frag(const bf16_t* base, int ldb, int lane) {
  const int r = lane & 15, h = (lane >> 4) & 1;
  const bf16_t* p = base + r * ldb + h * 8;
  v8bf lo = *(const v8bf*)(p);
  v8bf hi = *(const v8bf*)(p + 16);
  v16bf out;
#pragma unroll
  for (int i = 0; i < 8; ++i) { out[i] = lo[i]; out[8 + i] = hi[i]; }
  return out;
}

__device__ inline v8f wmma_bf16(v16bf a, v16bf b, v8f c) {
  return __builtin_amdgcn_wmma_f32_16x16x32_bf16(
      false, a, false, b, (short)0, c, false, false);
}

// ---------------------------------------------------------------------------
__global__ void cvt_f32_bf16(const float* __restrict__ in,
                             bf16_t* __restrict__ out, int n) {
  int i = blockIdx.x * blockDim.x + threadIdx.x;
  if (i < n) out[i] = (bf16_t)in[i];
}

// ---------------------------------------------------------------------------
// NT GEMM: out[m,n] = sum_k A[m,k] * Bw[n,k]
// Block 256 thr (8 waves), tile 128x128, K-step 32, double-buffered LDS fed
// by async global->LDS. Wave tile 32x64 (2 A-frags x 4 B-frags, 8 WMMA/step).
//   MODE 0: fp32 row-major [M, C_]
//   MODE 1: bf16 at [(b*H+h)*L + l]*D + d   (Q / K layout)
//   MODE 2: bf16 at [(b*H+h)*D + d]*L + l   (V transposed layout)
// ---------------------------------------------------------------------------
template <int MODE>
__global__ __launch_bounds__(256) void gemm_nt(const bf16_t* __restrict__ A,
                                               const bf16_t* __restrict__ Bw,
                                               void* __restrict__ out, int K) {
  __shared__ __align__(16) bf16_t sA[2][128 * 32];   // 2 x 8 KB
  __shared__ __align__(16) bf16_t sB[2][128 * 32];   // 2 x 8 KB

  const int t = threadIdx.x;
  const int lane = t & 31;
  const int w = t >> 5;
  const int wm = w & 3;         // 0..3 -> 32-row slab
  const int wn = w >> 2;        // 0..1 -> 64-col slab
  const int m_blk = blockIdx.x * 128;
  const int n_blk = blockIdx.y * 128;

  v8f acc[2][4] = {};

  const int srow = t >> 1, scol = (t & 1) * 16;  // 16 elems (2 x b128) per tile

  auto stage = [&](int buf, int kb) {
    const bf16_t* ga = A + (size_t)(m_blk + srow) * K + kb + scol;
    async_g2l_b128(&sA[buf][srow * 32 + scol],     ga);
    async_g2l_b128(&sA[buf][srow * 32 + scol + 8], ga + 8);
    const bf16_t* gb = Bw + (size_t)(n_blk + srow) * K + kb + scol;
    async_g2l_b128(&sB[buf][srow * 32 + scol],     gb);
    async_g2l_b128(&sB[buf][srow * 32 + scol + 8], gb + 8);
  };

  stage(0, 0);
  wait_async0();
  __syncthreads();

  for (int kb = 0; kb < K; kb += 32) {
    const int cur = (kb >> 5) & 1;
    if (kb + 32 < K) stage(cur ^ 1, kb + 32);   // DMA next tile during compute

    v16bf af0 = ld_a_frag(sA[cur] + (wm * 32) * 32, 32, lane);
    v16bf af1 = ld_a_frag(sA[cur] + (wm * 32 + 16) * 32, 32, lane);
#pragma unroll
    for (int j = 0; j < 4; ++j) {
      v16bf bfr = ld_b_frag(sB[cur] + (wn * 64 + j * 16) * 32, 32, lane);
      acc[0][j] = wmma_bf16(af0, bfr, acc[0][j]);
      acc[1][j] = wmma_bf16(af1, bfr, acc[1][j]);
    }
    wait_async0();
    __syncthreads();
  }

  const int r = lane & 15, hl = (lane >> 4) & 1;
#pragma unroll
  for (int i = 0; i < 2; ++i)
#pragma unroll
    for (int j = 0; j < 4; ++j)
#pragma unroll
      for (int v = 0; v < 8; ++v) {
        const int m = m_blk + wm * 32 + i * 16 + v + 8 * hl;
        const int n = n_blk + wn * 64 + j * 16 + r;
        const float val = acc[i][j][v];
        if (MODE == 0) {
          ((float*)out)[(size_t)m * C_ + n] = val;
        } else {
          const int b  = m >> 11;          // / L_
          const int l  = m & (L_ - 1);
          const int hd = n >> 6;           // / D_
          const int d  = n & (D_ - 1);
          if (MODE == 1)
            ((bf16_t*)out)[((size_t)(b * H_ + hd) * L_ + l) * D_ + d] = (bf16_t)val;
          else
            ((bf16_t*)out)[((size_t)(b * H_ + hd) * D_ + d) * L_ + l] = (bf16_t)val;
        }
      }
}

// ---------------------------------------------------------------------------
// Flash attention: per block one (b, head) and 128 q rows (wave w owns 16).
// Double-buffered async K/V tiles; one barrier per 64-key iteration.
// ---------------------------------------------------------------------------
__global__ __launch_bounds__(256) void flash_attn(const bf16_t* __restrict__ Q,
                                                  const bf16_t* __restrict__ Kg,
                                                  const bf16_t* __restrict__ Vt,
                                                  const int* __restrict__ kpm,
                                                  bf16_t* __restrict__ Y) {
  __shared__ __align__(16) bf16_t sQ[128 * 64];        // 16 KB
  __shared__ __align__(16) bf16_t sK[2][64 * 64];      // 2 x 8 KB
  __shared__ __align__(16) bf16_t sV[2][64 * 64];      // 2 x 8 KB (Vt: d x key)
  __shared__ __align__(16) bf16_t sP[8 * 16 * 64];     // 16 KB (per-wave P)
  __shared__ float sMask[2][64];

  const int t = threadIdx.x, lane = t & 31, w = t >> 5;
  const int q0 = blockIdx.x * 128;
  const int head = blockIdx.y;
  const int b = blockIdx.z;
  const size_t bh = (size_t)(b * H_ + head);

  auto stage_kv = [&](int buf, int kv) {
    const int row = t >> 2, co = (t & 3) * 16;
    const bf16_t* gk = Kg + (bh * L_ + kv + row) * D_ + co;
    async_g2l_b128(&sK[buf][row * 64 + co],     gk);
    async_g2l_b128(&sK[buf][row * 64 + co + 8], gk + 8);
    const bf16_t* gv = Vt + (bh * D_ + row) * L_ + kv + co;
    async_g2l_b128(&sV[buf][row * 64 + co],     gv);
    async_g2l_b128(&sV[buf][row * 64 + co + 8], gv + 8);
    if (t < 64) sMask[buf][t] = kpm[b * L_ + kv + t] ? 0.f : -__builtin_inff();
  };

  { // Q tile via async DMA: 4 x b128 per thread
    const int row = t >> 1, co = (t & 1) * 32;
    const bf16_t* g = Q + (bh * L_ + q0 + row) * D_ + co;
#pragma unroll
    for (int i = 0; i < 4; ++i)
      async_g2l_b128(&sQ[row * 64 + co + i * 8], g + i * 8);
  }
  stage_kv(0, 0);
  wait_async0();
  __syncthreads();

  const int r = lane & 15, hl = (lane >> 4) & 1;
  float m_run[8], l_run[8];
  v8f oacc[4] = {};
#pragma unroll
  for (int v = 0; v < 8; ++v) { m_run[v] = -__builtin_inff(); l_run[v] = 0.f; }
  const float scale = 0.125f;  // 1/sqrt(64)

  for (int kv = 0; kv < L_; kv += 64) {
    const int cur = (kv >> 6) & 1;
    if (kv + 64 < L_) stage_kv(cur ^ 1, kv + 64);   // DMA next tile

    // S = Q K^T  (16 q-rows x 64 keys per wave)
    v8f sf[4] = {};
    const bf16_t* aq = sQ + (w * 16) * 64;
    v16bf a0 = ld_a_frag(aq, 64, lane);
    v16bf a1 = ld_a_frag(aq + 32, 64, lane);
#pragma unroll
    for (int j = 0; j < 4; ++j) {
      v16bf b0 = ld_b_frag(sK[cur] + (j * 16) * 64, 64, lane);
      v16bf b1 = ld_b_frag(sK[cur] + (j * 16) * 64 + 32, 64, lane);
      sf[j] = wmma_bf16(a0, b0, sf[j]);
      sf[j] = wmma_bf16(a1, b1, sf[j]);
    }

    // scale + key padding mask
    float sc[4][8];
#pragma unroll
    for (int j = 0; j < 4; ++j)
#pragma unroll
      for (int v = 0; v < 8; ++v)
        sc[j][v] = sf[j][v] * scale + sMask[cur][j * 16 + r];

    // online softmax (row = v + 8*hl; reduce across 16-lane half)
    float alpha[8];
#pragma unroll
    for (int v = 0; v < 8; ++v) {
      float mx = fmaxf(fmaxf(sc[0][v], sc[1][v]), fmaxf(sc[2][v], sc[3][v]));
#pragma unroll
      for (int xs = 1; xs < 16; xs <<= 1) mx = fmaxf(mx, __shfl_xor(mx, xs, 32));
      const float mnew = fmaxf(m_run[v], mx);
      alpha[v] = (mnew == -__builtin_inff()) ? 1.0f : __expf(m_run[v] - mnew);
      m_run[v] = mnew;
      float rs = 0.f;
#pragma unroll
      for (int j = 0; j < 4; ++j) {
        const float p = __expf(sc[j][v] - mnew);
        rs += p;
        sP[w * 1024 + (v + 8 * hl) * 64 + j * 16 + r] = (bf16_t)p;
      }
#pragma unroll
      for (int xs = 1; xs < 16; xs <<= 1) rs += __shfl_xor(rs, xs, 32);
      l_run[v] = l_run[v] * alpha[v] + rs;
    }
#pragma unroll
    for (int j = 0; j < 4; ++j)
#pragma unroll
      for (int v = 0; v < 8; ++v) oacc[j][v] *= alpha[v];

    // wave-local P in LDS: drain the b16 stores before re-reading as frags
    asm volatile("s_wait_dscnt 0" ::: "memory");

    // O += P * Vt   (NT form: O[m,d] = sum_k P[m,k] * Vt[d,k])
    const bf16_t* ap = sP + w * 1024;
    v16bf p0 = ld_a_frag(ap, 64, lane);
    v16bf p1 = ld_a_frag(ap + 32, 64, lane);
#pragma unroll
    for (int j = 0; j < 4; ++j) {
      v16bf b0 = ld_b_frag(sV[cur] + (j * 16) * 64, 64, lane);
      v16bf b1 = ld_b_frag(sV[cur] + (j * 16) * 64 + 32, 64, lane);
      oacc[j] = wmma_bf16(p0, b0, oacc[j]);
      oacc[j] = wmma_bf16(p1, b1, oacc[j]);
    }
    wait_async0();
    __syncthreads();
  }

  // normalize and write y[b, l, head*64 + d] (bf16, [B,L,C] layout)
#pragma unroll
  for (int j = 0; j < 4; ++j)
#pragma unroll
    for (int v = 0; v < 8; ++v) {
      const float inv = 1.0f / l_run[v];
      const int l = q0 + w * 16 + v + 8 * hl;
      const int c = head * D_ + j * 16 + r;
      Y[((size_t)(b * L_ + l)) * C_ + c] = (bf16_t)(oacc[j][v] * inv);
    }
}

// ---------------------------------------------------------------------------
extern "C" void kernel_launch(void* const* d_in, const int* in_sizes, int n_in,
                              void* d_out, int out_size, void* d_ws, size_t ws_size,
                              hipStream_t stream) {
  (void)in_sizes; (void)n_in; (void)out_size; (void)ws_size;
  const float* x   = (const float*)d_in[0];
  const int*   kpm = (const int*)d_in[1];
  const float* Wq  = (const float*)d_in[2];
  const float* Wk  = (const float*)d_in[3];
  const float* Wv  = (const float*)d_in[4];
  const float* Wp  = (const float*)d_in[5];
  float* out = (float*)d_out;

  bf16_t* xb  = (bf16_t*)d_ws;                        // 4096*1024
  bf16_t* wq  = xb  + (size_t)M_ * C_;
  bf16_t* wk  = wq  + (size_t)C_ * C_;
  bf16_t* wv  = wk  + (size_t)C_ * C_;
  bf16_t* wp  = wv  + (size_t)C_ * C_;
  bf16_t* qb  = wp  + (size_t)C_ * C_;                // [B,H,L,D]
  bf16_t* kb  = qb  + (size_t)B_ * H_ * L_ * D_;      // [B,H,L,D]
  bf16_t* vtb = kb  + (size_t)B_ * H_ * L_ * D_;      // [B,H,D,L]
  bf16_t* yb  = vtb + (size_t)B_ * H_ * L_ * D_;      // [B,L,C]

  const int nx = M_ * C_, nw = C_ * C_;
  cvt_f32_bf16<<<(nx + 255) / 256, 256, 0, stream>>>(x, xb, nx);
  cvt_f32_bf16<<<(nw + 255) / 256, 256, 0, stream>>>(Wq, wq, nw);
  cvt_f32_bf16<<<(nw + 255) / 256, 256, 0, stream>>>(Wk, wk, nw);
  cvt_f32_bf16<<<(nw + 255) / 256, 256, 0, stream>>>(Wv, wv, nw);
  cvt_f32_bf16<<<(nw + 255) / 256, 256, 0, stream>>>(Wp, wp, nw);

  dim3 gg(M_ / 128, C_ / 128);
  gemm_nt<1><<<gg, 256, 0, stream>>>(xb, wq, qb, C_);
  gemm_nt<1><<<gg, 256, 0, stream>>>(xb, wk, kb, C_);
  gemm_nt<2><<<gg, 256, 0, stream>>>(xb, wv, vtb, C_);

  dim3 ga(L_ / 128, H_, B_);
  flash_attn<<<ga, 256, 0, stream>>>(qb, kb, vtb, kpm, yb);

  gemm_nt<0><<<gg, 256, 0, stream>>>(yb, wp, out, C_);
}
